// BahdanauAttention_28286654612080
// MI455X (gfx1250) — matrix operand order repaired and checked by reading, not compiled, not run on hardware
//
#include <hip/hip_runtime.h>
#include <hip/hip_bf16.h>
#include <math.h>

// Problem dims (fixed by reference)
#define B_  32
#define T_  2
#define S_  2048
#define H_  1024
#define TILE_S 64
// LDS row stride for the A tile: 1024 bf16 = 2048B, +16B pad so consecutive
// rows land 4 banks apart -> conflict-free ds_load_b128 across 16 lanes.
#define A_STRIDE 2064

typedef __bf16 bf16_t;
typedef __attribute__((ext_vector_type(16))) __bf16 v16bf;
typedef __attribute__((ext_vector_type(8)))  __bf16 v8bf;
typedef __attribute__((ext_vector_type(8)))  float  v8f;

static __device__ __forceinline__ unsigned short f2bf(float f) {
    unsigned u = __float_as_uint(f);
    unsigned r = u + 0x7FFFu + ((u >> 16) & 1u);   // round-to-nearest-even
    return (unsigned short)(r >> 16);
}

static __device__ __forceinline__ float fast_tanh(float x) {
    // tanh(x) = 1 - 2/(e^{2x}+1); safe at +/-inf (exp->inf gives 1, exp->0 gives -1)
    return 1.0f - 2.0f / (__expf(2.0f * x) + 1.0f);
}

// ---------------------------------------------------------------------------
// Kernel 1: qW1b[b,h] = (q[b] . W1[h,:]) + W1_b[h] + W2_b[h]   (tiny, 67 MFLOP)
// ---------------------------------------------------------------------------
__global__ void qw1_kernel(const float* __restrict__ query,
                           const float* __restrict__ W1,
                           const float* __restrict__ W1b,
                           const float* __restrict__ W2b,
                           float* __restrict__ qw1b) {
    __shared__ float qs[H_];
    const int b = blockIdx.x, t = threadIdx.x;
    const float* qrow = query + ((size_t)b * T_ + (T_ - 1)) * H_;   // last timestep
    for (int i = t; i < H_; i += 256) qs[i] = qrow[i];
    __syncthreads();
    for (int h = t; h < H_; h += 256) {
        const float4* wr = (const float4*)(W1 + (size_t)h * H_);
        const float4* qv = (const float4*)qs;
        float acc = 0.f;
#pragma unroll 4
        for (int k = 0; k < H_ / 4; ++k) {
            float4 a = wr[k], q = qv[k];
            acc += a.x * q.x + a.y * q.y + a.z * q.z + a.w * q.w;
        }
        qw1b[b * H_ + h] = acc + W1b[h] + W2b[h];
    }
}

// ---------------------------------------------------------------------------
// Kernel 2: W2 fp32 -> bf16 (2 MB result, L2-resident for the whole GEMM)
// ---------------------------------------------------------------------------
__global__ void w2cvt_kernel(const float* __restrict__ W2,
                             unsigned short* __restrict__ w2bf) {
    const int i = (blockIdx.x * 256 + threadIdx.x) * 4;
    float4 v = *(const float4*)(W2 + i);
    unsigned r0 = (unsigned)f2bf(v.x) | ((unsigned)f2bf(v.y) << 16);
    unsigned r1 = (unsigned)f2bf(v.z) | ((unsigned)f2bf(v.w) << 16);
    *(uint2*)(w2bf + i) = make_uint2(r0, r1);
}

// ---------------------------------------------------------------------------
// Kernel 3: the fused big GEMM + tanh + V_w reduction -> raw scores.
// One workgroup = one (b, 64-row S-block). 8 waves: wave w -> M-tile w&3,
// N-tile parity w>>2. 8192 x v_wmma_f32_16x16x32_bf16 per workgroup.
// ---------------------------------------------------------------------------
__global__ void __launch_bounds__(256, 1)
attn_scores_kernel(const float* __restrict__ values,
                   const unsigned short* __restrict__ w2bf,
                   const float* __restrict__ qw1b,
                   const float* __restrict__ Vw,
                   const float* __restrict__ Vb,
                   float* __restrict__ scores) {
    extern __shared__ __align__(16) char smem[];   // 64 x A_STRIDE bytes
    __shared__ float scoreAcc[TILE_S];

    const int t  = threadIdx.x;
    const int b  = blockIdx.x >> 5;
    const int s0 = (blockIdx.x & 31) * TILE_S;

    if (t < TILE_S) scoreAcc[t] = 0.f;

    // ---- Stage A tile: values[b, s0:s0+64, :] fp32 -> bf16 in LDS ----
    {
        const float* src = values + ((size_t)b * S_ + s0) * H_;
        for (int idx = t * 4; idx < TILE_S * H_; idx += 256 * 4) {
            const int row = idx >> 10, k = idx & (H_ - 1);
            float4 v = *(const float4*)(src + (size_t)row * H_ + k);
            unsigned r0 = (unsigned)f2bf(v.x) | ((unsigned)f2bf(v.y) << 16);
            unsigned r1 = (unsigned)f2bf(v.z) | ((unsigned)f2bf(v.w) << 16);
            *(uint2*)(smem + row * A_STRIDE + k * 2) = make_uint2(r0, r1);
        }
    }
    __syncthreads();

    const int w = t >> 5, l = t & 31;
    const int lane16 = l & 15, half = l >> 4;
    const int m0   = (w & 3) * 16;     // M tile within the 64 rows
    const int npar = w >> 2;           // N-tile parity (0/1)

    // A fragment base: row m0+lane16, K-chunk start 8*half (bytes: 16*half)
    const char* aBase = smem + (m0 + lane16) * A_STRIDE + 16 * half;

    float part[8];
#pragma unroll
    for (int v = 0; v < 8; ++v) part[v] = 0.f;

    for (int g = 0; g < 8; ++g) {
        v8f acc[4];
        const unsigned short* bp[4];
        int ncol[4];
#pragma unroll
        for (int j = 0; j < 4; ++j) {
            v8f z = {0.f, 0.f, 0.f, 0.f, 0.f, 0.f, 0.f, 0.f};
            acc[j] = z;
            const int nt = npar + 2 * (4 * g + j);      // covers nt = 0..63
            ncol[j] = nt * 16 + lane16;
            // B(K,N) element = W2[n,k]: row-major W2 row is contiguous in K.
            bp[j] = w2bf + ((size_t)ncol[j] * H_ + 8 * half);
        }

#pragma unroll 2
        for (int k0 = 0; k0 < H_; k0 += 32) {
            v8bf alo = *(const v8bf*)(aBase + 2 * k0);        // K = k0+8h .. +7
            v8bf ahi = *(const v8bf*)(aBase + 2 * k0 + 32);   // K = k0+8h+16 .. +23
            v16bf a = __builtin_shufflevector(alo, ahi,
                        0, 1, 2, 3, 4, 5, 6, 7, 8, 9, 10, 11, 12, 13, 14, 15);
#pragma unroll
            for (int j = 0; j < 4; ++j) {
                v8bf blo = *(const v8bf*)(bp[j] + k0);        // K = k0+8h .. +7
                v8bf bhi = *(const v8bf*)(bp[j] + k0 + 16);   // K = k0+8h+16 .. +23
                v16bf bf = __builtin_shufflevector(blo, bhi,
                        0, 1, 2, 3, 4, 5, 6, 7, 8, 9, 10, 11, 12, 13, 14, 15);
                acc[j] = __builtin_amdgcn_wmma_f32_16x16x32_bf16(
                             false, a, false, bf, (short)0, acc[j], false, false);
            }
        }

        // ---- fused epilogue: bias + tanh + V_w dot, accumulate per-row ----
#pragma unroll
        for (int j = 0; j < 4; ++j) {
            const float bias = qw1b[b * H_ + ncol[j]];
            const float vw   = Vw[ncol[j]];
#pragma unroll
            for (int v = 0; v < 8; ++v) {                    // C: M = m0+8*half+v
                part[v] += vw * fast_tanh(acc[j][v] + bias);
            }
        }
    }

    // Cross-lane reduce over the 16 lanes of each half (N dimension), then
    // combine the two N-parity waves per M-tile via LDS atomics.
#pragma unroll
    for (int v = 0; v < 8; ++v) {
        float p = part[v];
        p += __shfl_xor(p, 1, 32);
        p += __shfl_xor(p, 2, 32);
        p += __shfl_xor(p, 4, 32);
        p += __shfl_xor(p, 8, 32);
        if (lane16 == 0) atomicAdd(&scoreAcc[m0 + 8 * half + v], p);
    }
    __syncthreads();
    if (t < TILE_S) scores[b * S_ + s0 + t] = scoreAcc[t] + Vb[0];
}

// ---------------------------------------------------------------------------
// Kernel 4: masked softmax over S, in place in the alphas slot of d_out.
// ---------------------------------------------------------------------------
__global__ void softmax_kernel(const int* __restrict__ mask,
                               float* __restrict__ alphas) {
    __shared__ float sv[S_];
    __shared__ float red[256];
    const int b = blockIdx.x, t = threadIdx.x;
    float* sc = alphas + b * S_;
    const int* mb = mask + b * S_;

    float lm = -INFINITY;
    for (int s = t; s < S_; s += 256) {
        float v = (mb[s] == 0) ? -INFINITY : sc[s];
        sv[s] = v;
        lm = fmaxf(lm, v);
    }
    red[t] = lm; __syncthreads();
    for (int o = 128; o > 0; o >>= 1) {
        if (t < o) red[t] = fmaxf(red[t], red[t + o]);
        __syncthreads();
    }
    const float m = red[0]; __syncthreads();

    float ls = 0.f;
    for (int s = t; s < S_; s += 256) {
        float e = __expf(sv[s] - m);
        sv[s] = e; ls += e;
    }
    red[t] = ls; __syncthreads();
    for (int o = 128; o > 0; o >>= 1) {
        if (t < o) red[t] += red[t + o];
        __syncthreads();
    }
    const float inv = 1.f / red[0];
    for (int s = t; s < S_; s += 256) sc[s] = sv[s] * inv;
}

// ---------------------------------------------------------------------------
// Kernel 5: context[b,h] = sum_s alpha[b,s] * values[b,s,h] (memory bound)
// ---------------------------------------------------------------------------
__global__ void context_kernel(const float* __restrict__ values,
                               const float* __restrict__ alphas,
                               float* __restrict__ ctx) {
    const int b = blockIdx.y;
    const int h = blockIdx.x * 128 + threadIdx.x;
    const float* vb = values + (size_t)b * S_ * H_ + h;
    const float* al = alphas + b * S_;
    float acc = 0.f;
#pragma unroll 4
    for (int s = 0; s < S_; ++s) acc += al[s] * vb[(size_t)s * H_];
    ctx[b * H_ + h] = acc;
}

// ---------------------------------------------------------------------------
extern "C" void kernel_launch(void* const* d_in, const int* in_sizes, int n_in,
                              void* d_out, int out_size, void* d_ws, size_t ws_size,
                              hipStream_t stream) {
    const float* query  = (const float*)d_in[0];
    const float* values = (const float*)d_in[1];
    const int*   mask   = (const int*)  d_in[2];
    const float* W1w    = (const float*)d_in[3];
    const float* W1b    = (const float*)d_in[4];
    const float* W2w    = (const float*)d_in[5];
    const float* W2b    = (const float*)d_in[6];
    const float* Vw     = (const float*)d_in[7];
    const float* Vb     = (const float*)d_in[8];

    float* out    = (float*)d_out;
    float* ctx    = out;                 // (B,1,H)  = 32768 floats
    float* alphas = out + B_ * H_;       // (B,1,S)  = 65536 floats (scores staged here)

    float*          qw1b = (float*)d_ws;                                  // 128 KB
    unsigned short* w2bf = (unsigned short*)((char*)d_ws + B_ * H_ * 4);  // 2 MB

    qw1_kernel<<<B_, 256, 0, stream>>>(query, W1w, W1b, W2b, qw1b);
    w2cvt_kernel<<<(H_ * H_) / (256 * 4), 256, 0, stream>>>(W2w, w2bf);

    const size_t smem = (size_t)TILE_S * A_STRIDE;   // 132096 B (< 320 KB WGP LDS)
    (void)hipFuncSetAttribute((const void*)attn_scores_kernel,
                              hipFuncAttributeMaxDynamicSharedMemorySize, (int)smem);
    attn_scores_kernel<<<B_ * (S_ / TILE_S), 256, smem, stream>>>(
        values, w2bf, qw1b, Vw, Vb, alphas);

    softmax_kernel<<<B_, 256, 0, stream>>>(mask, alphas);
    context_kernel<<<dim3(H_ / 128, B_), 128, 0, stream>>>(values, alphas, ctx);
}
